// Encoder_50663434223728
// MI455X (gfx1250) — compile-verified, matrix operand order checked by reading, not run on hardware
//
#include <hip/hip_runtime.h>

#define SEQ 8192
#define EMB 2048
#define HID 2048
#define WROW (EMB + HID)   // 4096: W row = [input part | hidden part]

typedef __bf16 bf16;
typedef __attribute__((ext_vector_type(16))) __bf16 v16bf;
typedef __attribute__((ext_vector_type(8)))  __bf16 v8bf;
typedef __attribute__((ext_vector_type(8)))  float  v8f;
typedef __attribute__((ext_vector_type(4)))  float  v4f;

// ---------------------------------------------------------------------------
// Phase 1: input-projection GEMMs with bf16 WMMA, f32 accumulate.
// C[t][n] = sum_k X[t][k] * W[n][k]  (k = 0..EMB-1, input half of W) + bias[n]
// Result stored as bf16 in workspace.
// grid = (HID/128, SEQ/128, 3), block = 256 (8 waves)
// K-chunk = 64 per LDS fill (16 WMMAs per barrier pair).
// ---------------------------------------------------------------------------

__device__ __forceinline__ void cvt16_store(const float* __restrict__ src,
                                            bf16* __restrict__ dst) {
    v4f f0 = *(const v4f*)(src + 0);
    v4f f1 = *(const v4f*)(src + 4);
    v4f f2 = *(const v4f*)(src + 8);
    v4f f3 = *(const v4f*)(src + 12);
    v8bf lo, hi;
    lo[0] = (bf16)f0[0]; lo[1] = (bf16)f0[1]; lo[2] = (bf16)f0[2]; lo[3] = (bf16)f0[3];
    lo[4] = (bf16)f1[0]; lo[5] = (bf16)f1[1]; lo[6] = (bf16)f1[2]; lo[7] = (bf16)f1[3];
    hi[0] = (bf16)f2[0]; hi[1] = (bf16)f2[1]; hi[2] = (bf16)f2[2]; hi[3] = (bf16)f2[3];
    hi[4] = (bf16)f3[0]; hi[5] = (bf16)f3[1]; hi[6] = (bf16)f3[2]; hi[7] = (bf16)f3[3];
    *(v8bf*)(dst + 0) = lo;
    *(v8bf*)(dst + 8) = hi;
}

__global__ __launch_bounds__(256)
void gru_input_gemm(const float* __restrict__ X,
                    const float* __restrict__ Wz, const float* __restrict__ Wr,
                    const float* __restrict__ Wh,
                    const float* __restrict__ bz, const float* __restrict__ br,
                    const float* __restrict__ bh,
                    bf16* __restrict__ Xz, bf16* __restrict__ Xr,
                    bf16* __restrict__ Xh) {
    const int g = blockIdx.z;
    const float* W    = (g == 0) ? Wz : (g == 1) ? Wr : Wh;
    const float* bias = (g == 0) ? bz : (g == 1) ? br : bh;
    bf16* out         = (g == 0) ? Xz : (g == 1) ? Xr : Xh;

    const int tileN = blockIdx.x * 128;
    const int tileM = blockIdx.y * 128;

    constexpr int KT = 64;  // K-chunk per LDS fill
    constexpr int SA = 80;  // padded k-stride (bf16): 160B rows keep 32B alignment
    __shared__ __align__(32) bf16 As[128 * SA];  // 20KB
    __shared__ __align__(32) bf16 Bs[128 * SA];  // 20KB

    const int tid   = threadIdx.x;
    const int lane  = tid & 31;
    const int wave  = tid >> 5;          // 0..7
    const int waveM = (wave >> 2) * 64;  // 0 or 64
    const int waveN = (wave & 3) * 32;   // 0,32,64,96

    v8f acc[4][2];
#pragma unroll
    for (int i = 0; i < 4; ++i)
#pragma unroll
        for (int j = 0; j < 2; ++j) acc[i][j] = v8f{};

    // cooperative loader: 128 rows x 64 floats per tile; thread covers 16 floats,
    // two rows (stride 64) per tile.
    const int lr = tid >> 2;         // 0..63
    const int lc = (tid & 3) * 16;   // 0,16,32,48

    for (int kt = 0; kt < EMB; kt += KT) {
#pragma unroll
        for (int rr = 0; rr < 128; rr += 64) {
            cvt16_store(X + (size_t)(tileM + lr + rr) * EMB + (kt + lc),
                        &As[(lr + rr) * SA + lc]);
            cvt16_store(W + (size_t)(tileN + lr + rr) * WROW + (kt + lc),
                        &Bs[(lr + rr) * SA + lc]);
        }
        __syncthreads();

        const int half = (lane >> 4) * 16;
#pragma unroll
        for (int kk = 0; kk < KT; kk += 32) {
            v16bf a[4], b[2];
#pragma unroll
            for (int mi = 0; mi < 4; ++mi)
                a[mi] = *(const v16bf*)&As[(waveM + mi * 16 + (lane & 15)) * SA + kk + half];
#pragma unroll
            for (int ni = 0; ni < 2; ++ni)
                b[ni] = *(const v16bf*)&Bs[(waveN + ni * 16 + (lane & 15)) * SA + kk + half];
#pragma unroll
            for (int mi = 0; mi < 4; ++mi)
#pragma unroll
                for (int ni = 0; ni < 2; ++ni)
                    acc[mi][ni] = __builtin_amdgcn_wmma_f32_16x16x32_bf16(
                        false, a[mi], false, b[ni], (short)0, acc[mi][ni], false, false);
        }
        __syncthreads();
    }

    // epilogue: C layout — VGPR r: lanes 0-15 => (M = 8*(lane>>4)+r, N = lane&15)
    const int chalf = lane >> 4;
    const int ccol  = lane & 15;
#pragma unroll
    for (int ni = 0; ni < 2; ++ni) {
        const int col = tileN + waveN + ni * 16 + ccol;
        const float bv = bias[col];
#pragma unroll
        for (int mi = 0; mi < 4; ++mi) {
            const int rowbase = tileM + waveM + mi * 16 + chalf * 8;
#pragma unroll
            for (int r = 0; r < 8; ++r)
                out[(size_t)(rowbase + r) * HID + col] = (bf16)(acc[mi][ni][r] + bv);
        }
    }
}

// ---------------------------------------------------------------------------
// Phase 1b: hidden-half weights -> bf16, transposed AND packed:
//   PW[kb][n][j] = W[n][EMB + kb*8 + j]   (kb = k/8, j = k%8)
// so each recurrence lane reads one b128 (8 bf16 of consecutive k) per kb.
// grid = (32, 32, 3), block = 256
// ---------------------------------------------------------------------------
__global__ __launch_bounds__(256)
void gru_w_pack(const float* __restrict__ Wz, const float* __restrict__ Wr,
                const float* __restrict__ Wh,
                bf16* __restrict__ WzT, bf16* __restrict__ WrT,
                bf16* __restrict__ WhT) {
    const int g = blockIdx.z;
    const float* W = (g == 0) ? Wz : (g == 1) ? Wr : Wh;
    bf16* WT       = (g == 0) ? WzT : (g == 1) ? WrT : WhT;

    __shared__ float tile[64][65];   // [nLocal][kLocal]
    const int n0 = blockIdx.x * 64;
    const int k0 = blockIdx.y * 64;
    const int tx = threadIdx.x & 63;
    const int ty = threadIdx.x >> 6;  // 0..3

#pragma unroll
    for (int i = 0; i < 64; i += 4)
        tile[ty + i][tx] = W[(size_t)(n0 + ty + i) * WROW + EMB + k0 + tx];
    __syncthreads();

    // pack: thread -> (nLocal = tx, kb = ty + p), writes v8bf; lanes contiguous in n
    const int kb0 = k0 >> 3;
#pragma unroll
    for (int p = 0; p < 8; p += 4) {
        const int kb = ty + p;  // local kb 0..7
        v8bf v;
#pragma unroll
        for (int j = 0; j < 8; ++j) v[j] = (bf16)tile[tx][kb * 8 + j];
        *(v8bf*)&WT[(((size_t)(kb0 + kb)) * HID + n0 + tx) * 8] = v;
    }
}

// ---------------------------------------------------------------------------
// Phase 2: persistent sequential GRU recurrence.
// 32 WGs x 512 threads. WG owns 64 outputs; 8 k-chunks of 256 per output.
// Two device-wide barriers per step (r-vector dependency).
// ---------------------------------------------------------------------------
#define RWG 32
#define RTH 512
#define RNL 64     // outputs per WG
#define RKC 8      // k-chunks
#define RKL 256    // k per chunk (= 32 packed groups of 8)

__device__ __forceinline__ void gbar(int* cnt, int* gen, int target) {
    __threadfence();  // make this thread's prior global stores device-visible
    __syncthreads();
    if (threadIdx.x == 0) {
        if (atomicAdd(cnt, 1) == RWG - 1) {
            atomicExch(cnt, 0);
            __hip_atomic_store(gen, target, __ATOMIC_RELEASE, __HIP_MEMORY_SCOPE_AGENT);
        } else {
            while (__hip_atomic_load(gen, __ATOMIC_ACQUIRE, __HIP_MEMORY_SCOPE_AGENT) < target)
                __builtin_amdgcn_s_sleep(1);
        }
    }
    __syncthreads();
}

__global__ __launch_bounds__(RTH)
void gru_recurrence(const bf16* __restrict__ Xz, const bf16* __restrict__ Xr,
                    const bf16* __restrict__ Xh,
                    const bf16* __restrict__ WzT, const bf16* __restrict__ WrT,
                    const bf16* __restrict__ WhT,
                    float* __restrict__ hbuf0, float* __restrict__ hbuf1,
                    float* __restrict__ rvec,
                    int* __restrict__ bar_cnt, int* __restrict__ bar_gen,
                    float* __restrict__ out) {
    __shared__ float hs[HID];    // current h
    __shared__ float rhs[HID];   // r * h
    __shared__ float red[RTH];   // reduction scratch

    const int tid = threadIdx.x;
    const int nl  = tid & (RNL - 1);          // local output index
    const int kc  = tid >> 6;                 // k-chunk 0..7
    const int n   = blockIdx.x * RNL + nl;    // global hidden unit
    const int k0  = kc * RKL;

    // packed-weight base for this thread: PW[(kb)*HID + n][8]
    const size_t wbase = (((size_t)(k0 >> 3)) * HID + n) * 8;
    const size_t wstep = (size_t)HID * 8;

    float* hb[2] = {hbuf0, hbuf1};

    for (int t = 0; t < SEQ; ++t) {
        const float* hcur = hb[t & 1];
        float* hnext      = hb[(t + 1) & 1];

        for (int i = tid; i < HID; i += RTH) hs[i] = hcur[i];
        __syncthreads();

        // ---- phase A: z and r gates ----
        float sz = 0.f, sr = 0.f;
        {
            const bf16* wz = WzT + wbase;
            const bf16* wr = WrT + wbase;
#pragma unroll 4
            for (int kb = 0; kb < RKL / 8; ++kb) {
                const v8bf wzv = *(const v8bf*)(wz + (size_t)kb * wstep);
                const v8bf wrv = *(const v8bf*)(wr + (size_t)kb * wstep);
#pragma unroll
                for (int j = 0; j < 8; ++j) {
                    const float hv = hs[k0 + kb * 8 + j];
                    sz = fmaf((float)wzv[j], hv, sz);
                    sr = fmaf((float)wrv[j], hv, sr);
                }
            }
        }
        red[tid] = sz;
        __syncthreads();
        float z = 0.f, r = 0.f, hold = 0.f;
        if (kc == 0) {
            sz = 0.f;
#pragma unroll
            for (int j = 0; j < RKC; ++j) sz += red[nl + RNL * j];
        }
        __syncthreads();
        red[tid] = sr;
        __syncthreads();
        if (kc == 0) {
            sr = 0.f;
#pragma unroll
            for (int j = 0; j < RKC; ++j) sr += red[nl + RNL * j];
            z = 1.f / (1.f + expf(-(sz + (float)Xz[(size_t)t * HID + n])));
            r = 1.f / (1.f + expf(-(sr + (float)Xr[(size_t)t * HID + n])));
            rvec[n] = r;
            hold    = hs[n];
        }
        gbar(bar_cnt, bar_gen, 2 * t + 1);

        // ---- phase B: candidate + h update ----
        for (int i = tid; i < HID; i += RTH) rhs[i] = rvec[i] * hs[i];
        __syncthreads();
        float sh = 0.f;
        {
            const bf16* wh = WhT + wbase;
#pragma unroll 4
            for (int kb = 0; kb < RKL / 8; ++kb) {
                const v8bf whv = *(const v8bf*)(wh + (size_t)kb * wstep);
#pragma unroll
                for (int j = 0; j < 8; ++j)
                    sh = fmaf((float)whv[j], rhs[k0 + kb * 8 + j], sh);
            }
        }
        red[tid] = sh;
        __syncthreads();
        if (kc == 0) {
            sh = 0.f;
#pragma unroll
            for (int j = 0; j < RKC; ++j) sh += red[nl + RNL * j];
            const float cand = tanhf(sh + (float)Xh[(size_t)t * HID + n]);
            const float hn   = (1.f - z) * hold + z * cand;
            out[(size_t)t * HID + n] = hn;
            hnext[n] = hn;
            if (t == SEQ - 1) out[(size_t)SEQ * HID + n] = hn;  // h_final
        }
        gbar(bar_cnt, bar_gen, 2 * t + 2);
    }
}

// ---------------------------------------------------------------------------
extern "C" void kernel_launch(void* const* d_in, const int* in_sizes, int n_in,
                              void* d_out, int out_size, void* d_ws, size_t ws_size,
                              hipStream_t stream) {
    const float* X  = (const float*)d_in[0];
    const float* Wz = (const float*)d_in[1];
    const float* Wr = (const float*)d_in[2];
    const float* Wh = (const float*)d_in[3];
    const float* bz = (const float*)d_in[4];
    const float* br = (const float*)d_in[5];
    const float* bh = (const float*)d_in[6];
    float* out = (float*)d_out;

    char* ws = (char*)d_ws;
    const size_t MB = 1024u * 1024u;
    bf16* Xz  = (bf16*)(ws + 0 * MB);     // SEQ*HID bf16 = 32MB
    bf16* Xr  = (bf16*)(ws + 32 * MB);
    bf16* Xh  = (bf16*)(ws + 64 * MB);
    bf16* WzT = (bf16*)(ws + 96 * MB);    // HID*HID bf16 = 8MB (packed [k/8][n][8])
    bf16* WrT = (bf16*)(ws + 104 * MB);
    bf16* WhT = (bf16*)(ws + 112 * MB);
    float* hbuf0 = (float*)(ws + 120 * MB);
    float* hbuf1 = hbuf0 + HID;
    float* rvec  = hbuf1 + HID;
    int*   bar   = (int*)(rvec + HID);    // [cnt, gen]

    // zero h0, h1, rvec, barrier state (graph-capture safe)
    hipMemsetAsync(hbuf0, 0, (size_t)3 * HID * sizeof(float) + 2 * sizeof(int), stream);

    dim3 gGemm(HID / 128, SEQ / 128, 3);
    gru_input_gemm<<<gGemm, 256, 0, stream>>>(X, Wz, Wr, Wh, bz, br, bh, Xz, Xr, Xh);

    dim3 gTr(HID / 64, HID / 64, 3);
    gru_w_pack<<<gTr, 256, 0, stream>>>(Wz, Wr, Wh, WzT, WrT, WhT);

    gru_recurrence<<<RWG, RTH, 0, stream>>>(Xz, Xr, Xh, WzT, WrT, WhT,
                                            hbuf0, hbuf1, rvec, bar, bar + 1, out);
}
